// Block_69535520522345
// MI455X (gfx1250) — compile-verified, matrix operand order checked by reading, not run on hardware
//
#include <hip/hip_runtime.h>
#include <hip/hip_bf16.h>
#include <math.h>

// Problem sizes (fixed by the reference)
#define BB 2
#define TT 2048
#define CC 1024
#define HH 16
#define HD 64
#define MM (BB*TT)   // 4096 rows

typedef __attribute__((ext_vector_type(16))) __bf16 v16bf;
typedef __attribute__((ext_vector_type(8)))  __bf16 v8bf;
typedef __attribute__((ext_vector_type(8)))  float  v8f;

union Frag16 { v16bf v; v8bf h[2]; __bf16 e[16]; };

__device__ __forceinline__ __bf16 f2bf(float f) { return (__bf16)f; }

__device__ __forceinline__ v8f wmma_bf16(const Frag16& a, const Frag16& b, v8f c) {
  // D = A(16x32 bf16) * B(32x16 bf16) + C(16x16 f32)
  return __builtin_amdgcn_wmma_f32_16x16x32_bf16(
      /*neg_a=*/false, a.v, /*neg_b=*/false, b.v,
      /*c_mod=*/(short)0, c, /*reuse_a=*/false, /*reuse_b=*/false);
}

// Async copy 16 bytes global -> LDS (ASYNCcnt-tracked, no VGPR data).
// Per-lane: each lane moves its own 16B. LDS address = low 32 bits of the
// generic shared-aperture address.
__device__ __forceinline__ void async_ld16(const __bf16* g, __bf16* l) {
  unsigned lds = (unsigned)(size_t)l;
  unsigned long long ga = (unsigned long long)(size_t)g;
  asm volatile("global_load_async_to_lds_b128 %0, %1, off"
               :: "v"(lds), "v"(ga) : "memory");
}

// ---------------------------------------------------------------- fp32 -> bf16
__global__ void cvt_f32_bf16(const float* __restrict__ in,
                             __bf16* __restrict__ out, int n) {
  int i = blockIdx.x * blockDim.x + threadIdx.x;
  if (i < n) out[i] = (__bf16)in[i];
}

// ---------------------------------------------------------------- LayerNorm
__global__ __launch_bounds__(256)
void layernorm_bf16(const float* __restrict__ x, const float* __restrict__ w,
                    const float* __restrict__ b, __bf16* __restrict__ out) {
  __shared__ float red[256];
  int row = blockIdx.x;
  int tid = threadIdx.x;
  const float* xr = x + (size_t)row * CC;
  float vals[4];
  float s = 0.f, s2 = 0.f;
  for (int i = 0; i < 4; i++) {
    float v = xr[tid + i * 256];
    vals[i] = v; s += v; s2 += v * v;
  }
  red[tid] = s; __syncthreads();
  for (int st = 128; st > 0; st >>= 1) {
    if (tid < st) red[tid] += red[tid + st];
    __syncthreads();
  }
  float mean = red[0] * (1.0f / CC);
  __syncthreads();
  red[tid] = s2; __syncthreads();
  for (int st = 128; st > 0; st >>= 1) {
    if (tid < st) red[tid] += red[tid + st];
    __syncthreads();
  }
  float var = red[0] * (1.0f / CC) - mean * mean;
  float rinv = rsqrtf(var + 1e-5f);
  for (int i = 0; i < 4; i++) {
    int c = tid + i * 256;
    out[(size_t)row * CC + c] = f2bf((vals[i] - mean) * rinv * w[c] + b[c]);
  }
}

// ---------------------------------------------------------------- WMMA GEMM
// out[M,N] = act(A[M,K] @ W[N,K]^T + bias) (+ resid); A,W bf16; fp32 acc.
// Block: 256 thr (8 waves), tile 128x128, BK=32, double-buffered async-LDS.
// Wave -> 32x64 of output: 8 independent WMMAs per K-step.
#define GBM 128
#define GBN 128
#define GBK 32
__global__ __launch_bounds__(256)
void gemm_wmma(const __bf16* __restrict__ A, const __bf16* __restrict__ W,
               const float* __restrict__ bias, const float* __restrict__ resid,
               float* __restrict__ outF, __bf16* __restrict__ outB,
               int M, int N, int K, int act) {
  __shared__ __bf16 As[2][GBM][GBK];   // 2 x 8KB
  __shared__ __bf16 Bs[2][GBN][GBK];   // 2 x 8KB
  int tid = threadIdx.x;
  int wid = tid >> 5, lane = tid & 31;
  int bm = blockIdx.y * GBM, bn = blockIdx.x * GBN;
  int mR = (wid & 3) * 32;             // wave M-slice (4 slices)
  int nR = (wid >> 2) * 64;            // wave N-slice (2 slices)
  int lrow = lane & 15;
  int k0 = (lane < 16) ? 0 : 8;        // fragment K-swizzle
  int rowOff = (lane >= 16) ? 8 : 0;

  // Staging plan: 512 x 16B chunks per matrix tile, 2 chunks/thread each.
  // chunk -> row = chunk>>2, col = (chunk&3)*8   (4 chunks per 32-elem row)
  auto stage = [&](int kk, int bufq) {
    for (int i = 0; i < 2; i++) {
      int chunk = tid + 256 * i;
      int row = chunk >> 2, c8 = (chunk & 3) * 8;
      async_ld16(&A[(size_t)(bm + row) * K + kk + c8], &As[bufq][row][c8]);
    }
    for (int i = 0; i < 2; i++) {
      int chunk = tid + 256 * i;
      int row = chunk >> 2, c8 = (chunk & 3) * 8;
      async_ld16(&W[(size_t)(bn + row) * K + kk + c8], &Bs[bufq][row][c8]);
    }
  };

  v8f acc[2][4] = {};
  stage(0, 0);
  int buf = 0;
  for (int kk = 0; kk < K; kk += GBK) {
    bool more = (kk + GBK) < K;
    if (more) {
      stage(kk + GBK, buf ^ 1);                       // overlap with compute
      asm volatile("s_wait_asynccnt 0x4" ::: "memory"); // current tile done
    } else {
      asm volatile("s_wait_asynccnt 0x0" ::: "memory");
    }
    __syncthreads();
    Frag16 a0, a1;
    a0.h[0] = *(const v8bf*)&As[buf][mR + lrow][k0];
    a0.h[1] = *(const v8bf*)&As[buf][mR + lrow][k0 + 16];
    a1.h[0] = *(const v8bf*)&As[buf][mR + 16 + lrow][k0];
    a1.h[1] = *(const v8bf*)&As[buf][mR + 16 + lrow][k0 + 16];
    for (int j = 0; j < 4; j++) {
      Frag16 bfr;
      bfr.h[0] = *(const v8bf*)&Bs[buf][nR + j * 16 + lrow][k0];
      bfr.h[1] = *(const v8bf*)&Bs[buf][nR + j * 16 + lrow][k0 + 16];
      acc[0][j] = wmma_bf16(a0, bfr, acc[0][j]);
      acc[1][j] = wmma_bf16(a1, bfr, acc[1][j]);
    }
    __syncthreads();
    buf ^= 1;
  }

  for (int i = 0; i < 2; i++) {
    for (int j = 0; j < 4; j++) {
      for (int r = 0; r < 8; r++) {
        int m = bm + mR + i * 16 + r + rowOff;
        int n = bn + nR + j * 16 + lrow;
        float v = acc[i][j][r] + bias[n];
        if (act == 1) {
          float u = v;
          v = 0.5f * u * (1.0f + tanhf(0.7978845608028654f * (u + 0.044715f * u * u * u)));
        }
        size_t idx = (size_t)m * N + n;
        if (resid) v += resid[idx];
        if (outF) outF[idx] = v;
        if (outB) outB[idx] = f2bf(v);
      }
    }
  }
}

// ---------------------------------------------------------------- split QKV
__global__ void pack_qkv(const float* __restrict__ qkv,
                         __bf16* __restrict__ qh, __bf16* __restrict__ kh,
                         __bf16* __restrict__ vh, float* __restrict__ present) {
  int i = blockIdx.x * blockDim.x + threadIdx.x;   // over B*T*C
  if (i >= BB * TT * CC) return;
  int c = i % CC;
  int t = (i / CC) % TT;
  int b = i / (CC * TT);
  int h = c / HD, d = c % HD;
  size_t base = (size_t)(b * TT + t) * (3 * CC);
  float q = qkv[base + c];
  float k = qkv[base + CC + c];
  float v = qkv[base + 2 * CC + c];
  size_t hidx = ((size_t)(b * HH + h) * TT + t) * HD + d;
  qh[hidx] = f2bf(q); kh[hidx] = f2bf(k); vh[hidx] = f2bf(v);
  present[hidx] = k;
  present[(size_t)BB * HH * TT * HD + hidx] = v;
}

// ---------------------------------------------------------------- flash attn
// Block: 256 thr = 8 waves; each wave owns 16 query rows of one (b,h).
// Streaming softmax with entropy state: H = m + log(l) - t/l.
// V tiles staged to LDS via async copies that overlap the QK^T WMMAs.
__global__ __launch_bounds__(256)
void attn_wmma(const __bf16* __restrict__ qh, const __bf16* __restrict__ kh,
               const __bf16* __restrict__ vh, __bf16* __restrict__ y,
               float* __restrict__ entOut) {
  __shared__ __bf16 Pls[8][16][32];   // per-wave P tile (re-swizzle buffer)
  __shared__ __bf16 Vls[8][32][64];   // per-wave V tile staging
  int tid = threadIdx.x, wid = tid >> 5, lane = tid & 31;
  int h = blockIdx.y, b = blockIdx.z;
  int m0 = blockIdx.x * 128 + wid * 16;        // first query row of this wave
  size_t hb = ((size_t)(b * HH + h)) * TT * HD;
  const __bf16* qb = qh + hb;
  const __bf16* kb = kh + hb;
  const __bf16* vb = vh + hb;
  int lrow = lane & 15;
  int k0 = (lane < 16) ? 0 : 8;
  int rowOff = (lane >= 16) ? 8 : 0;
  const float NEG_INF = -__builtin_inff();

  // Q fragments (16x64 -> two 16x32 A-fragments), kept in VGPRs
  Frag16 qf[2];
  {
    size_t qoff = (size_t)(m0 + lrow) * HD;
    for (int j = 0; j < 2; j++) {
      qf[j].h[0] = *(const v8bf*)&qb[qoff + j * 32 + k0];
      qf[j].h[1] = *(const v8bf*)&qb[qoff + j * 32 + k0 + 16];
    }
  }

  float mrow[8], lsum[8], tsum[8];
  v8f o[4] = {};
  for (int r = 0; r < 8; r++) { mrow[r] = NEG_INF; lsum[r] = 0.f; tsum[r] = 0.f; }

  int nTiles = (m0 + 47) >> 5;                 // cover keys 0..m0+15, step 32
  for (int kt = 0; kt < nTiles; kt++) {
    int kbase = kt * 32;
    // All prior ds reads of Vls must drain before async writes reuse it
    // (async ops are unordered vs DS ops).
    asm volatile("s_wait_dscnt 0x0" ::: "memory");
    // stage V tile 32x64 async (lane copies one key row, clamped);
    // overlaps with the QK^T WMMAs below.
    {
      int vrow = kbase + lane; if (vrow > TT - 1) vrow = TT - 1;
      const __bf16* vr = vb + (size_t)vrow * HD;
      for (int j = 0; j < 8; j++)
        async_ld16(&vr[j * 8], &Vls[wid][lane][j * 8]);
    }
    // S tiles: two 16(q)x16(k) accumulators, K chained over HD=64
    v8f s0acc = {}; v8f s1acc = {};
    {
      int kr0 = kbase + lrow;      if (kr0 > TT - 1) kr0 = TT - 1;
      int kr1 = kbase + 16 + lrow; if (kr1 > TT - 1) kr1 = TT - 1;
      size_t koff0 = (size_t)kr0 * HD, koff1 = (size_t)kr1 * HD;
      Frag16 kf;
      kf.h[0] = *(const v8bf*)&kb[koff0 + k0];
      kf.h[1] = *(const v8bf*)&kb[koff0 + k0 + 16];
      s0acc = wmma_bf16(qf[0], kf, s0acc);
      kf.h[0] = *(const v8bf*)&kb[koff0 + 32 + k0];
      kf.h[1] = *(const v8bf*)&kb[koff0 + 32 + k0 + 16];
      s0acc = wmma_bf16(qf[1], kf, s0acc);
      kf.h[0] = *(const v8bf*)&kb[koff1 + k0];
      kf.h[1] = *(const v8bf*)&kb[koff1 + k0 + 16];
      s1acc = wmma_bf16(qf[0], kf, s1acc);
      kf.h[0] = *(const v8bf*)&kb[koff1 + 32 + k0];
      kf.h[1] = *(const v8bf*)&kb[koff1 + 32 + k0 + 16];
      s1acc = wmma_bf16(qf[1], kf, s1acc);
    }
    // streaming softmax + entropy state; P written to LDS in bf16
    for (int r = 0; r < 8; r++) {
      int rowq = m0 + r + rowOff;
      int key0 = kbase + lrow, key1 = kbase + 16 + lrow;
      bool msk0 = key0 > rowq, msk1 = key1 > rowq;
      float s0 = s0acc[r] * 0.125f;            // 1/sqrt(64)
      float s1 = s1acc[r] * 0.125f;
      float lm = fmaxf(msk0 ? NEG_INF : s0, msk1 ? NEG_INF : s1);
      for (int xm = 1; xm < 16; xm <<= 1) lm = fmaxf(lm, __shfl_xor(lm, xm, 32));
      float mnew = fmaxf(mrow[r], lm);
      float sc = __expf(mrow[r] - mnew);
      float e0 = msk0 ? 0.f : __expf(s0 - mnew);
      float e1 = msk1 ? 0.f : __expf(s1 - mnew);
      float ls = e0 + e1;
      float ts = (msk0 ? 0.f : e0 * s0) + (msk1 ? 0.f : e1 * s1);
      for (int xm = 1; xm < 16; xm <<= 1) {
        ls += __shfl_xor(ls, xm, 32);
        ts += __shfl_xor(ts, xm, 32);
      }
      lsum[r] = lsum[r] * sc + ls;
      tsum[r] = tsum[r] * sc + ts;
      mrow[r] = mnew;
      for (int dg = 0; dg < 4; dg++) o[dg][r] *= sc;
      int prow = r + rowOff;
      Pls[wid][prow][lrow]      = f2bf(e0);
      Pls[wid][prow][16 + lrow] = f2bf(e1);
    }
    // wave-local ordering: P (ds) and V (async) must both be resident
    asm volatile("s_wait_dscnt 0x0" ::: "memory");
    asm volatile("s_wait_asynccnt 0x0" ::: "memory");
    // P as A-fragment (16x32)
    Frag16 pf;
    pf.h[0] = *(const v8bf*)&Pls[wid][lrow][k0];
    pf.h[1] = *(const v8bf*)&Pls[wid][lrow][k0 + 16];
    // V as B-fragments (32x16 per 16-wide d-group), O += P @ V
    for (int dg = 0; dg < 4; dg++) {
      Frag16 vf;
      int d = dg * 16 + lrow;
      for (int e = 0; e < 8; e++) vf.e[e]     = Vls[wid][k0 + e][d];
      for (int e = 0; e < 8; e++) vf.e[8 + e] = Vls[wid][k0 + 16 + e][d];
      o[dg] = wmma_bf16(pf, vf, o[dg]);
    }
  }
  // epilogue: normalize O, entropy, write y in (B,T,C) merged-head layout
  float entSum = 0.f;
  for (int r = 0; r < 8; r++) {
    float invl = 1.0f / lsum[r];
    entSum += mrow[r] + __logf(lsum[r]) - tsum[r] * invl;
    int q = m0 + r + rowOff;
    for (int dg = 0; dg < 4; dg++) {
      int d = dg * 16 + lrow;
      y[((size_t)(b * TT + q)) * CC + h * HD + d] = f2bf(o[dg][r] * invl);
    }
  }
  if (lrow == 0)
    atomicAdd(entOut, entSum * (1.0f / (float)(BB * HH * TT)));
}

// ---------------------------------------------------------------- launch
extern "C" void kernel_launch(void* const* d_in, const int* in_sizes, int n_in,
                              void* d_out, int out_size, void* d_ws, size_t ws_size,
                              hipStream_t stream) {
  (void)in_sizes; (void)n_in; (void)out_size; (void)ws_size;
  const float* x      = (const float*)d_in[0];
  const float* ln1_w  = (const float*)d_in[1];
  const float* ln1_b  = (const float*)d_in[2];
  const float* w_attn = (const float*)d_in[3];
  const float* b_attn = (const float*)d_in[4];
  const float* w_proj = (const float*)d_in[5];
  const float* b_proj = (const float*)d_in[6];
  const float* ln2_w  = (const float*)d_in[7];
  const float* ln2_b  = (const float*)d_in[8];
  const float* w_fc   = (const float*)d_in[9];
  const float* b_fc   = (const float*)d_in[10];
  const float* w_mlp  = (const float*)d_in[11];
  const float* b_mlp  = (const float*)d_in[12];

  float* outX       = (float*)d_out;
  float* outEnt     = outX + (size_t)MM * CC;
  float* outPresent = outEnt + 1;

  char* wsp = (char*)d_ws;
  size_t off = 0;
  auto alloc = [&](size_t bytes) -> void* {
    void* p = wsp + off;
    off += (bytes + 255) & ~(size_t)255;
    return p;
  };
  __bf16* hbf   = (__bf16*)alloc((size_t)MM * CC * 2);
  float*  qkv   = (float*) alloc((size_t)MM * 3 * CC * 4);
  __bf16* qhB   = (__bf16*)alloc((size_t)MM * CC * 2);
  __bf16* khB   = (__bf16*)alloc((size_t)MM * CC * 2);
  __bf16* vhB   = (__bf16*)alloc((size_t)MM * CC * 2);
  __bf16* ybf   = (__bf16*)alloc((size_t)MM * CC * 2);
  float*  x1    = (float*) alloc((size_t)MM * CC * 4);
  __bf16* h2bf  = (__bf16*)alloc((size_t)MM * CC * 2);
  __bf16* m1bf  = (__bf16*)alloc((size_t)MM * 4 * CC * 2);
  __bf16* wAtB  = (__bf16*)alloc((size_t)3 * CC * CC * 2);
  __bf16* wPrB  = (__bf16*)alloc((size_t)CC * CC * 2);
  __bf16* wFcB  = (__bf16*)alloc((size_t)4 * CC * CC * 2);
  __bf16* wMlB  = (__bf16*)alloc((size_t)4 * CC * CC * 2);

  // weight conversion (fp32 -> bf16)
  cvt_f32_bf16<<<(3 * CC * CC + 255) / 256, 256, 0, stream>>>(w_attn, wAtB, 3 * CC * CC);
  cvt_f32_bf16<<<(CC * CC + 255) / 256, 256, 0, stream>>>(w_proj, wPrB, CC * CC);
  cvt_f32_bf16<<<(4 * CC * CC + 255) / 256, 256, 0, stream>>>(w_fc, wFcB, 4 * CC * CC);
  cvt_f32_bf16<<<(4 * CC * CC + 255) / 256, 256, 0, stream>>>(w_mlp, wMlB, 4 * CC * CC);

  // attention branch
  layernorm_bf16<<<MM, 256, 0, stream>>>(x, ln1_w, ln1_b, hbf);
  gemm_wmma<<<dim3(3 * CC / GBN, MM / GBM), 256, 0, stream>>>(
      hbf, wAtB, b_attn, nullptr, qkv, nullptr, MM, 3 * CC, CC, 0);
  pack_qkv<<<(MM * CC + 255) / 256, 256, 0, stream>>>(qkv, qhB, khB, vhB, outPresent);
  hipMemsetAsync(outEnt, 0, sizeof(float), stream);
  attn_wmma<<<dim3(TT / 128, HH, BB), 256, 0, stream>>>(qhB, khB, vhB, ybf, outEnt);
  gemm_wmma<<<dim3(CC / GBN, MM / GBM), 256, 0, stream>>>(
      ybf, wPrB, b_proj, x, x1, nullptr, MM, CC, CC, 0);

  // MLP branch
  layernorm_bf16<<<MM, 256, 0, stream>>>(x1, ln2_w, ln2_b, h2bf);
  gemm_wmma<<<dim3(4 * CC / GBN, MM / GBM), 256, 0, stream>>>(
      h2bf, wFcB, b_fc, nullptr, nullptr, m1bf, MM, 4 * CC, CC, 1);
  gemm_wmma<<<dim3(CC / GBN, MM / GBM), 256, 0, stream>>>(
      m1bf, wMlB, b_mlp, x1, outX, nullptr, MM, CC, 4 * CC, 0);
}
